// NeuralGDEForecaster_32177894981952
// MI455X (gfx1250) — compile-verified, matrix operand order checked by reading, not run on hardware
//
#include <hip/hip_runtime.h>
#include <hip/hip_bf16.h>
#include <math.h>

typedef __attribute__((ext_vector_type(16))) _Float16 v16h;
typedef __attribute__((ext_vector_type(8)))  float    v8f;

namespace {
constexpr int kB   = 4;
constexpr int kN   = 2000;
constexpr int kT   = 16;
constexpr int kFH  = 8;
constexpr int kE   = 32000;
constexpr int kTN  = kT * kN;                       // 32000
constexpr int kBTN = kB * kTN;                      // 128000
constexpr int kETOT = kT * kE + 2 * (kT - 1) * kN;  // 572000
constexpr int kNE  = kBTN * 64;                     // 8,192,000 elements in a big buffer
}

__device__ __forceinline__ void atomic_add_f32(float* p, float v) {
#if __has_builtin(__hip_atomic_fetch_add)
  __hip_atomic_fetch_add(p, v, __ATOMIC_RELAXED, __HIP_MEMORY_SCOPE_AGENT);
#else
  atomicAdd(p, v);
#endif
}

// K index inside a 16x32 f16 A/B fragment: lane group g (0/1), element j (0..15)
__device__ __forceinline__ int frag_k(int g, int j) {
  return (j < 8) ? (8 * g + j) : (16 + 8 * g + (j - 8));
}

// ---------------- setup kernels ----------------

__global__ void k_init_deg(float* deg, int n) {
  int i = blockIdx.x * blockDim.x + threadIdx.x;
  if (i < n) deg[i] = 1.0f;  // self-loop
}

__global__ void k_build_edges(const int* ei, int* src, int* dst, float* deg) {
  int e = blockIdx.x * blockDim.x + threadIdx.x;
  if (e >= kETOT) return;
  int s, d;
  if (e < kT * kE) {
    int t = e / kE, k = e % kE;
    s = ei[k] + t * kN;
    d = ei[kE + k] + t * kN;
  } else {
    int r = e - kT * kE;
    if (r < (kT - 1) * kN) { s = r;      d = r + kN; }
    else { int r2 = r - (kT - 1) * kN;   s = r2 + kN; d = r2; }
  }
  src[e] = s; dst[e] = d;
  atomic_add_f32(&deg[d], 1.0f);
}

__global__ void k_norm(float* dis, float* snorm, int n) {
  int i = blockIdx.x * blockDim.x + threadIdx.x;
  if (i >= n) return;
  float r = rsqrtf(dis[i]);   // dis[] held degree
  dis[i] = r;
  snorm[i] = r * r;
}

// Split a 64x64 f32 weight (out,in) into fragment-ordered f16 hi/lo tables.
// Slot t = ((c*2+s)*32 + lane)*16 + j holds W[n][k], n=16c+lane%16, k=32s+frag_k(g,j)
__global__ void k_wsplit(const float* W, _Float16* hi, _Float16* lo) {
  int t = blockIdx.x * blockDim.x + threadIdx.x;
  if (t >= 4096) return;
  int j = t & 15, lane = (t >> 4) & 31, cs = t >> 9;
  int s = cs & 1, c = cs >> 1, g = lane >> 4;
  int n = 16 * c + (lane & 15);
  int k = 32 * s + frag_k(g, j);
  float v = W[n * 64 + k];
  _Float16 h = (_Float16)v;
  hi[t] = h;
  lo[t] = (_Float16)(v - (float)h);
}

// ---------------- split-f16 WMMA GEMM: out = in @ W^T ----------------
// in = c0*in0 + c1*in1 + c2*in2 + c3*in3 (first nt terms), rows = kBTN, 64->64.
// MODE 0 (GCN): hw = gemm; out = gemm*snorm[row%TN] + bias
// MODE 1 (attention): out = tanh(gemm + bias)
template <int MODE>
__global__ __launch_bounds__(256) void k_gemm(
    const float* in0, const float* in1, const float* in2, const float* in3,
    float c0, float c1, float c2, float c3, int nt,
    const _Float16* wfrag, const float* bias, const float* snorm,
    float* hw, float* out) {
  int wave = (blockIdx.x * blockDim.x + threadIdx.x) >> 5;
  int lane = threadIdx.x & 31;
  int row0 = wave * 16;
  if (row0 >= kBTN) return;                 // uniform per wave
  int m = lane & 15, g = lane >> 4;
  size_t rb = (size_t)(row0 + m) * 64;

  v16h ahi[2], alo[2];
#pragma unroll
  for (int s = 0; s < 2; ++s) {
#pragma unroll
    for (int j = 0; j < 16; ++j) {
      int k = 32 * s + frag_k(g, j);
      float v = c0 * in0[rb + k];
      if (nt > 1) v += c1 * in1[rb + k];
      if (nt > 2) v += c2 * in2[rb + k];
      if (nt > 3) v += c3 * in3[rb + k];
      _Float16 h = (_Float16)v;
      ahi[s][j] = h;
      alo[s][j] = (_Float16)(v - (float)h);
    }
  }

  // Rows this lane's D elements land on: row0 + 8*g + v (v = 0..7).
  // kTN is a multiple of 16, so flat = row % kTN is tile-uniform + small offset.
  int rbase = row0 + 8 * g;
  float sn[8];
  if (MODE == 0) {
    int fbase = (row0 % kTN) + 8 * g;   // one scalar mod per wave
#pragma unroll
    for (int v = 0; v < 8; ++v) sn[v] = snorm[fbase + v];
  }

  const _Float16* whi = wfrag;
  const _Float16* wlo = wfrag + 4096;
#pragma unroll
  for (int c = 0; c < 4; ++c) {
    v16h bh0 = *(const v16h*)(whi + ((size_t)(c * 2 + 0) * 32 + lane) * 16);
    v16h bh1 = *(const v16h*)(whi + ((size_t)(c * 2 + 1) * 32 + lane) * 16);
    v16h bl0 = *(const v16h*)(wlo + ((size_t)(c * 2 + 0) * 32 + lane) * 16);
    v16h bl1 = *(const v16h*)(wlo + ((size_t)(c * 2 + 1) * 32 + lane) * 16);
    v8f acc = {};
    acc = __builtin_amdgcn_wmma_f32_16x16x32_f16(false, ahi[0], false, bh0, (short)0, acc, false, false);
    acc = __builtin_amdgcn_wmma_f32_16x16x32_f16(false, ahi[1], false, bh1, (short)0, acc, false, false);
    acc = __builtin_amdgcn_wmma_f32_16x16x32_f16(false, alo[0], false, bh0, (short)0, acc, false, false);
    acc = __builtin_amdgcn_wmma_f32_16x16x32_f16(false, alo[1], false, bh1, (short)0, acc, false, false);
    acc = __builtin_amdgcn_wmma_f32_16x16x32_f16(false, ahi[0], false, bl0, (short)0, acc, false, false);
    acc = __builtin_amdgcn_wmma_f32_16x16x32_f16(false, ahi[1], false, bl1, (short)0, acc, false, false);
    int col = 16 * c + m;
    float bcol = bias[col];
#pragma unroll
    for (int v = 0; v < 8; ++v) {
      size_t o = (size_t)(rbase + v) * 64 + col;
      float val = acc[v];
      if (MODE == 0) {
        hw[o] = val;
        out[o] = val * sn[v] + bcol;
      } else {
        out[o] = tanhf(val + bcol);
      }
    }
  }
}

// ---------------- edge scatter-add: out[dst] += hw[src] * dis[s]*dis[d] ----------------
// grid.y = batch, grid.x covers kETOT*16 (each thread: 4 floats of one edge)
__global__ void k_edge(const int* src, const int* dst, const float* dis,
                       const float* hw, float* out) {
  int t = blockIdx.x * blockDim.x + threadIdx.x;
  if (t >= kETOT * 16) return;
  int quad = t & 15;
  int e = t >> 4;
  int b = blockIdx.y;
  int s = src[e], d = dst[e];
  float w = dis[s] * dis[d];
  const float* pin = hw + ((size_t)(b * kTN + s)) * 64 + quad * 4;
  float* pout = out + ((size_t)(b * kTN + d)) * 64 + quad * 4;
  atomic_add_f32(pout + 0, pin[0] * w);
  atomic_add_f32(pout + 1, pin[1] * w);
  atomic_add_f32(pout + 2, pin[2] * w);
  atomic_add_f32(pout + 3, pin[3] * w);
}

// ---------------- elementwise ----------------
__global__ void k_relu(float* p, int n) {
  int i = blockIdx.x * blockDim.x + threadIdx.x;
  if (i < n) p[i] = fmaxf(p[i], 0.0f);
}
__global__ void k_tanhk(float* p, int n) {
  int i = blockIdx.x * blockDim.x + threadIdx.x;
  if (i < n) p[i] = tanhf(p[i]);
}
__global__ void k_axpy3(float* y, const float* a, const float* b, const float* c,
                        float ca, float cb, float cc, int n) {
  int i = blockIdx.x * blockDim.x + threadIdx.x;
  if (i < n) y[i] += ca * a[i] + cb * b[i] + cc * c[i];
}
__global__ void k_axpy1(float* y, const float* a, float ca, int n) {
  int i = blockIdx.x * blockDim.x + threadIdx.x;
  if (i < n) y[i] += ca * a[i];
}

// ---------------- layer 0 (din = 1) ----------------
__global__ void k_layer0(const float* x, const float* Wg0, const float* bg0,
                         const float* snorm, float* hw, float* out) {
  int t = blockIdx.x * blockDim.x + threadIdx.x;
  if (t >= kNE) return;
  int j = t & 63, row = t >> 6;
  int b = row / kTN, flat = row % kTN;
  int tt = flat / kN, n = flat % kN;
  float xv = x[((size_t)(b * kN + n)) * kT + tt];
  float v = xv * Wg0[j];
  hw[t] = v;
  out[t] = v * snorm[flat] + bg0[j];
}

// ---------------- attention tail / GRU / head ----------------
__global__ void k_a3(const float* a2, const float* Wa3, const float* ba3, float* sa3) {
  int r = blockIdx.x * blockDim.x + threadIdx.x;
  if (r >= kBTN) return;
  const float* row = a2 + (size_t)r * 64;
  float s = ba3[0];
#pragma unroll
  for (int j = 0; j < 64; ++j) s += row[j] * Wa3[j];
  sa3[r] = s;
}

__global__ void k_softmax_t(float* sa3) {
  int bn = blockIdx.x * blockDim.x + threadIdx.x;
  if (bn >= kB * kN) return;
  int b = bn / kN, n = bn % kN;
  float v[kT], mx = -3.4e38f;
#pragma unroll
  for (int t = 0; t < kT; ++t) { v[t] = sa3[b * kTN + t * kN + n]; mx = fmaxf(mx, v[t]); }
  float sum = 0.f;
#pragma unroll
  for (int t = 0; t < kT; ++t) { v[t] = expf(v[t] - mx); sum += v[t]; }
  float inv = 1.0f / sum;
#pragma unroll
  for (int t = 0; t < kT; ++t) sa3[b * kTN + t * kN + n] = v[t] * inv;
}

__global__ void k_pool(const float* h, const float* w, float* nf) {
  int t = blockIdx.x * blockDim.x + threadIdx.x;
  if (t >= kB * kN * 64) return;
  int j = t & 63, bn = t >> 6;
  int b = bn / kN, n = bn % kN;
  float s = 0.f;
#pragma unroll
  for (int tt = 0; tt < kT; ++tt)
    s += h[((size_t)(b * kTN + tt * kN + n)) * 64 + j] * w[b * kTN + tt * kN + n];
  nf[t] = s;
}

__global__ void k_gru(const float* nf, const float* Wih, const float* bih,
                      const float* bhh, float* hid) {
  int t = blockIdx.x * blockDim.x + threadIdx.x;
  if (t >= kB * kN * 64) return;
  int j = t & 63, bn = t >> 6;
  const float* row = nf + (size_t)bn * 64;
  float rp = 0.f, zp = 0.f, np = 0.f;
#pragma unroll
  for (int k = 0; k < 64; ++k) {
    float v = row[k];
    rp += v * Wih[(size_t)j * 64 + k];
    zp += v * Wih[(size_t)(64 + j) * 64 + k];
    np += v * Wih[(size_t)(128 + j) * 64 + k];
  }
  rp += bih[j] + bhh[j];
  zp += bih[64 + j] + bhh[64 + j];
  float r = 1.0f / (1.0f + expf(-rp));
  float z = 1.0f / (1.0f + expf(-zp));
  float nc = tanhf(np + bih[128 + j] + r * bhh[128 + j]);
  hid[t] = (1.0f - z) * nc;
}

// y0[b, n*T + t, :] = hid[b, n, :]   (replicates reference's (b,n,t) flat layout)
__global__ void k_bcast(const float* hid, float* y) {
  int t = blockIdx.x * blockDim.x + threadIdx.x;
  if (t >= kNE) return;
  int j = t & 63, row = t >> 6;
  int b = row / kTN, rem = row % kTN, n = rem / kT;
  y[t] = hid[((size_t)(b * kN + n)) * 64 + j];
}

// pred[b, n, f] = dot(y[b, 15*N + n, :], Wout) + bout  (flat index as written in ref)
__global__ void k_pred(const float* y, const float* Wout, const float* bout,
                       float* pred, int f) {
  int bn = blockIdx.x * blockDim.x + threadIdx.x;
  if (bn >= kB * kN) return;
  int b = bn / kN, n = bn % kN;
  const float* row = y + ((size_t)(b * kTN + 15 * kN + n)) * 64;
  float s = bout[0];
#pragma unroll
  for (int j = 0; j < 64; ++j) s += row[j] * Wout[j];
  pred[(size_t)bn * kFH + f] = s;
}

// ---------------- host orchestration ----------------
extern "C" void kernel_launch(void* const* d_in, const int* in_sizes, int n_in,
                              void* d_out, int out_size, void* d_ws, size_t ws_size,
                              hipStream_t stream) {
  (void)in_sizes; (void)n_in; (void)out_size; (void)ws_size;
  const float* x   = (const float*)d_in[0];
  const int* eidx  = (const int*)d_in[1];
  const float* Wg0 = (const float*)d_in[2];  const float* bg0 = (const float*)d_in[3];
  const float* Wg1 = (const float*)d_in[4];  const float* bg1 = (const float*)d_in[5];
  const float* Wg2 = (const float*)d_in[6];  const float* bg2 = (const float*)d_in[7];
  const float* Wg3 = (const float*)d_in[8];  const float* bg3 = (const float*)d_in[9];
  const float* Wg4 = (const float*)d_in[10]; const float* bg4 = (const float*)d_in[11];
  const float* Wa1 = (const float*)d_in[12]; const float* ba1 = (const float*)d_in[13];
  const float* Wa2 = (const float*)d_in[14]; const float* ba2 = (const float*)d_in[15];
  const float* Wa3 = (const float*)d_in[16]; const float* ba3 = (const float*)d_in[17];
  const float* Wih = (const float*)d_in[18];
  const float* bih = (const float*)d_in[20]; const float* bhh = (const float*)d_in[21];
  const float* Wo0 = (const float*)d_in[22]; const float* bo0 = (const float*)d_in[23];
  const float* Wo1 = (const float*)d_in[24]; const float* bo1 = (const float*)d_in[25];
  const float* Wout = (const float*)d_in[26]; const float* bout = (const float*)d_in[27];
  float* pred = (float*)d_out;

  char* ws = (char*)d_ws;
  size_t off = 0;
  auto alloc = [&](size_t bytes) -> void* {
    void* p = ws + off;
    off += (bytes + 255) & ~(size_t)255;
    return p;
  };
  const size_t big = (size_t)kNE * sizeof(float);     // 32.768 MB each
  float* y  = (float*)alloc(big);
  float* k1 = (float*)alloc(big);
  float* k2 = (float*)alloc(big);
  float* k3 = (float*)alloc(big);
  float* hw = (float*)alloc(big);
  float* ob = (float*)alloc(big);
  int* src  = (int*)alloc((size_t)kETOT * 4);
  int* dst  = (int*)alloc((size_t)kETOT * 4);
  float* dis   = (float*)alloc((size_t)kTN * 4);
  float* snorm = (float*)alloc((size_t)kTN * 4);
  float* sa3 = (float*)alloc((size_t)kBTN * 4);
  float* nf  = (float*)alloc((size_t)kB * kN * 64 * 4);
  float* hid = (float*)alloc((size_t)kB * kN * 64 * 4);
  _Float16* wsp = (_Float16*)alloc((size_t)8 * 8192 * sizeof(_Float16));

  const int TPB = 256;
  const int gElem = (kNE + TPB - 1) / TPB;
  const int gGemm = kBTN / 16 / 8;                    // 1 wave per 16-row tile, 8 waves/block
  dim3 gEdge((kETOT * 16 + TPB - 1) / TPB, kB, 1);

  // ---- graph setup ----
  k_init_deg<<<(kTN + TPB - 1) / TPB, TPB, 0, stream>>>(dis, kTN);
  k_build_edges<<<(kETOT + TPB - 1) / TPB, TPB, 0, stream>>>(eidx, src, dst, dis);
  k_norm<<<(kTN + TPB - 1) / TPB, TPB, 0, stream>>>(dis, snorm, kTN);

  const float* wms[8] = {Wg1, Wg2, Wg3, Wg4, Wa1, Wa2, Wo0, Wo1};
  for (int m = 0; m < 8; ++m)
    k_wsplit<<<16, TPB, 0, stream>>>(wms[m], wsp + (size_t)m * 8192, wsp + (size_t)m * 8192 + 4096);
  const _Float16* sWg[4] = {wsp, wsp + 8192, wsp + 2 * 8192, wsp + 3 * 8192};
  const _Float16* sWa1 = wsp + 4 * 8192;
  const _Float16* sWa2 = wsp + 5 * 8192;
  const _Float16* sWo0 = wsp + 6 * 8192;
  const _Float16* sWo1 = wsp + 7 * 8192;

  auto gemm_gcn = [&](const float* i0, const float* i1, const float* i2, const float* i3,
                      float c0, float c1, float c2, float c3, int nt,
                      const _Float16* wf, const float* bias, float* outp) {
    k_gemm<0><<<gGemm, TPB, 0, stream>>>(i0, i1, i2, i3, c0, c1, c2, c3, nt,
                                         wf, bias, snorm, hw, outp);
  };
  auto gemm_act = [&](const float* i0, const _Float16* wf, const float* bias, float* outp) {
    k_gemm<1><<<gGemm, TPB, 0, stream>>>(i0, nullptr, nullptr, nullptr, 1.f, 0.f, 0.f, 0.f, 1,
                                         wf, bias, nullptr, nullptr, outp);
  };
  auto edges = [&](float* outp) {
    k_edge<<<gEdge, TPB, 0, stream>>>(src, dst, dis, hw, outp);
  };
  auto relu = [&](float* p) { k_relu<<<gElem, TPB, 0, stream>>>(p, kNE); };
  auto vtanh = [&](float* p) { k_tanhk<<<gElem, TPB, 0, stream>>>(p, kNE); };

  // ---- spatial GCN stack ----
  k_layer0<<<gElem, TPB, 0, stream>>>(x, Wg0, bg0, snorm, hw, ob);
  edges(ob); relu(ob);
  float* hcur = ob; float* hnext = y;
  const float* spb[4] = {bg1, bg2, bg3, bg4};
  for (int l = 0; l < 4; ++l) {
    gemm_gcn(hcur, nullptr, nullptr, nullptr, 1.f, 0.f, 0.f, 0.f, 1, sWg[l], spb[l], hnext);
    edges(hnext); relu(hnext);
    float* tp = hcur; hcur = hnext; hnext = tp;
  }
  // hcur == ob (final spatial features)

  // ---- temporal attention + GRU + broadcast ----
  gemm_act(hcur, sWa1, ba1, k1);
  gemm_act(k1, sWa2, ba2, k2);
  k_a3<<<(kBTN + TPB - 1) / TPB, TPB, 0, stream>>>(k2, Wa3, ba3, sa3);
  k_softmax_t<<<(kB * kN + TPB - 1) / TPB, TPB, 0, stream>>>(sa3);
  k_pool<<<(kB * kN * 64 + TPB - 1) / TPB, TPB, 0, stream>>>(hcur, sa3, nf);
  k_gru<<<(kB * kN * 64 + TPB - 1) / TPB, TPB, 0, stream>>>(nf, Wih, bih, bhh, hid);
  k_bcast<<<gElem, TPB, 0, stream>>>(hid, y);
  k_pred<<<(kB * kN + TPB - 1) / TPB, TPB, 0, stream>>>(y, Wout, bout, pred, 0);

  // ---- Neural ODE: RK4 (3/8 rule), 7 steps ----
  const float dtv = 8.0f / 7.0f;
  auto gcn_half = [&](const float* i0, const float* i1, const float* i2, const float* i3,
                      float c0, float c1, float c2, float c3, int nt,
                      const _Float16* wf, const float* bias, float* dstp) {
    gemm_gcn(i0, i1, i2, i3, c0, c1, c2, c3, nt, wf, bias, dstp);
    edges(dstp);
    vtanh(dstp);
  };
  auto feval = [&](const float* i0, const float* i1, const float* i2, const float* i3,
                   float c0, float c1, float c2, float c3, int nt, float* kdst) {
    gcn_half(i0, i1, i2, i3, c0, c1, c2, c3, nt, sWo0, bo0, ob);
    gcn_half(ob, nullptr, nullptr, nullptr, 1.f, 0.f, 0.f, 0.f, 1, sWo1, bo1, kdst);
  };
  for (int stepi = 0; stepi < kFH - 1; ++stepi) {
    feval(y, nullptr, nullptr, nullptr, 1.f, 0.f, 0.f, 0.f, 1, k1);
    feval(y, k1, nullptr, nullptr, 1.f, dtv / 3.f, 0.f, 0.f, 2, k2);
    feval(y, k1, k2, nullptr, 1.f, -dtv / 3.f, dtv, 0.f, 3, k3);
    // k4: first GCN reads (y,k1,k2,k3); then y no longer needed as an f-input
    gcn_half(y, k1, k2, k3, 1.f, dtv, -dtv, dtv, 4, sWo0, bo0, ob);
    k_axpy3<<<gElem, TPB, 0, stream>>>(y, k1, k2, k3,
                                       dtv * 0.125f, 3.f * dtv * 0.125f, 3.f * dtv * 0.125f, kNE);
    gcn_half(ob, nullptr, nullptr, nullptr, 1.f, 0.f, 0.f, 0.f, 1, sWo1, bo1, k1);  // k4 -> k1
    k_axpy1<<<gElem, TPB, 0, stream>>>(y, k1, dtv * 0.125f, kNE);
    k_pred<<<(kB * kN + TPB - 1) / TPB, TPB, 0, stream>>>(y, Wout, bout, pred, stepi + 1);
  }
}
